// MhAConv_40759239639258
// MI455X (gfx1250) — matrix operand order repaired and checked by reading, not compiled, not run on hardware
//
#include <hip/hip_runtime.h>
#include <hip/hip_bf16.h>

// ---- problem constants (from reference) ----
#define CIN   64
#define COUT  64
#define E_    32
#define HEAD  4
#define KS    3
#define IH    56
#define IW    56
#define MULT  (HEAD*KS)          // 12

// ---- tiling ----
#define TILE_H 7
#define TILE_W 14
#define HT_H   9                 // halo rows
#define HT_W   16                // halo cols
#define NPIX   (HT_H*HT_W)       // 144 halo pixels
#define NINT   (TILE_H*TILE_W)   // 98 interior pixels

typedef __attribute__((ext_vector_type(16))) __bf16 v16bf;
typedef __attribute__((ext_vector_type(8)))  __bf16 v8bf;
typedef __attribute__((ext_vector_type(8)))  float  v8f;

// ---- LDS layout (byte offsets, all 16B aligned) ----
#define Y1_STRIDE 33             // f32 words per pixel row (padded: conflict-free)
#define Y2_STRIDE 36             // f32 words per pixel row (padded + 16B aligned float4)
#define OFF_Y1   0                                   // 144*33*4 = 19008
#define OFF_B    19008                               // union: sX bf16 (18432) then sY2 f32 (20736)
#define OFF_WIN  (OFF_B + 20736)                     // 39744 : w_in  bf16 32x64
#define OFF_WOUT (OFF_WIN + 4096)                    // 43840 : w_out bf16 64x32
#define OFF_WQ   (OFF_WOUT + 4096)                   // 47936 : w_q f32 (13824B)
#define OFF_WK   (OFF_WQ + 13824)                    // 61760
#define OFF_WV   (OFF_WK + 13824)                    // 75584
#define OFF_WPE  (OFF_WV + 13824)                    // 89408 : w_pe f32 (1152B)
#define OFF_WP1  (OFF_WPE + 1152)                    // 90560 : w_p1 f32 (1536B)
#define SMEM_BYTES (OFF_WP1 + 1536)                  // 92096

__device__ __forceinline__ float bcastf(float v, int l) {
  return __int_as_float(__builtin_amdgcn_readlane(__float_as_int(v), l));
}

// Async DMA global->LDS, 16B per lane, tracked by ASYNCcnt.
__device__ __forceinline__ void async_b128(unsigned ldsByteOff, const void* g) {
  asm volatile("global_load_async_to_lds_b128 %0, %1, off"
               :: "v"(ldsByteOff), "v"((unsigned long long)(uintptr_t)g)
               : "memory");
}

__global__ __launch_bounds__(256, 1)
void MhAConv_fused_kernel(const float* __restrict__ x,
                          const float* __restrict__ w_in,
                          const float* __restrict__ w_q,
                          const float* __restrict__ w_k,
                          const float* __restrict__ w_v,
                          const float* __restrict__ w_pe,
                          const float* __restrict__ w_p1,
                          const float* __restrict__ w_out,
                          float* __restrict__ out) {
  __shared__ __align__(32) unsigned char smem[SMEM_BYTES];
  float*  sY1   = (float*)(smem + OFF_Y1);    // inProj output [pixel][e], stride 33
  __bf16* sX    = (__bf16*)(smem + OFF_B);    // x halo tile [pixel][cin] bf16
  float*  sY2   = (float*)(smem + OFF_B);     // proj1+pe out [pixel][e], stride 36 (reuses sX)
  __bf16* sWin  = (__bf16*)(smem + OFF_WIN);  // [e][cin]
  __bf16* sWout = (__bf16*)(smem + OFF_WOUT); // [o][e]
  const float* sWq  = (const float*)(smem + OFF_WQ);
  const float* sWk  = (const float*)(smem + OFF_WK);
  const float* sWv  = (const float*)(smem + OFF_WV);
  const float* sWpe = (const float*)(smem + OFF_WPE);
  const float* sWp1 = (const float*)(smem + OFF_WP1);

  const int tid  = threadIdx.x;
  const int lane = tid & 31;
  const int wave = tid >> 5;
  const int bx = blockIdx.x, by = blockIdx.y, bz = blockIdx.z;
  const unsigned sbase = (unsigned)(uintptr_t)(void*)smem;   // LDS byte offset of smem

  // ===== stage 0a: kick off async DMA of stage-2 weights into LDS (hidden behind
  // stage-0 staging + stage-1 WMMA; waited on just before stage 2). =====
  {
    // w_q/w_k/w_v: 3456 f32 = 864 x 16B chunks each; w_pe: 72 chunks; w_p1: 96 chunks
    for (int i = tid; i < 864; i += 256) {
      async_b128(sbase + OFF_WQ + 16u*i, (const char*)w_q + 16u*i);
      async_b128(sbase + OFF_WK + 16u*i, (const char*)w_k + 16u*i);
      async_b128(sbase + OFF_WV + 16u*i, (const char*)w_v + 16u*i);
    }
    for (int i = tid; i < 72;  i += 256)
      async_b128(sbase + OFF_WPE + 16u*i, (const char*)w_pe + 16u*i);
    for (int i = tid; i < 96;  i += 256)
      async_b128(sbase + OFF_WP1 + 16u*i, (const char*)w_p1 + 16u*i);
  }

  // ================= stage 0b: stage x halo + GEMM weights into LDS (bf16) =================
  if (tid < NPIX) {
    const int n  = tid;
    const int hy = n >> 4, hx = n & 15;
    const int gy = by * TILE_H + hy - 1;
    const int gx = bx * TILE_W + hx - 1;
    const bool inb = (gy >= 0) && (gy < IH) && (gx >= 0) && (gx < IW);
    const size_t base = (size_t)bz * CIN * (IH * IW);
    #pragma unroll 4
    for (int c = 0; c < CIN; ++c) {
      float v = inb ? x[base + (size_t)c * (IH * IW) + gy * IW + gx] : 0.0f;
      sX[n * CIN + c] = (__bf16)v;
    }
  }
  #pragma unroll
  for (int j = 0; j < 8; ++j) {               // 2048 elems each, 8 per thread
    int idx = tid * 8 + j;
    sWin[idx]  = (__bf16)w_in[idx];
    sWout[idx] = (__bf16)w_out[idx];
  }
  __syncthreads();

  // ================= stage 1: inProj GEMM  Y1[e][n] = Win[e][k] X[k][n] =================
  // M=32 (2 tiles) x N=144 (9 tiles) x K=64 (2 wmma steps), bf16 WMMA 16x16x32
  for (int t = wave; t < 18; t += 8) {
    const int Mt = t & 1;
    const int Nt = t >> 1;
    const int nPix = Nt * 16 + (lane & 15);
    v8f acc = {};
    #pragma unroll
    for (int kb = 0; kb < CIN; kb += 32) {
      // A fragment (16x32 bf16 layout): lanes 0-15 rows, 16 contiguous K from kb+16*(lane>>4)
      const int Mrow = Mt * 16 + (lane & 15);
      const int r0 = kb + 16 * (lane >> 4);
      v8bf alo = *(const v8bf*)(sWin + Mrow * CIN + r0);
      v8bf ahi = *(const v8bf*)(sWin + Mrow * CIN + r0 + 8);
      v16bf a = __builtin_shufflevector(alo, ahi, 0,1,2,3,4,5,6,7,8,9,10,11,12,13,14,15);
      // B fragment (32x16 bf16): lane column = pixel, 16 contiguous K in [pixel][cin]
      const int k0 = kb + 16 * (lane >> 4);
      v16bf b = *(const v16bf*)(sX + nPix * CIN + k0);
      acc = __builtin_amdgcn_wmma_f32_16x16x32_bf16(false, a, false, b, (short)0, acc, false, false);
    }
    #pragma unroll
    for (int r = 0; r < 8; ++r) {
      const int e = Mt * 16 + r + 8 * (lane >> 4);
      sY1[nPix * Y1_STRIDE + e] = acc[r];
    }
  }
  // all of this wave's async weight DMAs must have landed before any wave enters stage 2
  asm volatile("s_wait_asynccnt 0" ::: "memory");
  __syncthreads();

  // ========= stage 2: depthwise q/k/v convs + per-(pixel,head) attention =========
  // wave = pixel worker, lane = channel c (E=32 == wave32). k/v rows broadcast via v_readlane.
  {
    const int c = lane;
    float wpe[9];
    #pragma unroll
    for (int t9 = 0; t9 < 9; ++t9) wpe[t9] = sWpe[c * 9 + t9];

    for (int h = 0; h < HEAD; ++h) {
      float wq[3][9], wk[3][9], wv[3][9], wp[3];
      #pragma unroll
      for (int kk = 0; kk < 3; ++kk) {
        const int ch = (c * MULT + h * KS + kk) * 9;
        #pragma unroll
        for (int t9 = 0; t9 < 9; ++t9) {
          wq[kk][t9] = sWq[ch + t9];
          wk[kk][t9] = sWk[ch + t9];
          wv[kk][t9] = sWv[ch + t9];
        }
        wp[kk] = sWp1[c * MULT + h * KS + kk];
      }

      for (int i = wave; i < NINT; i += 8) {
        const int iy = i / TILE_W, ix = i % TILE_W;
        const int n  = (iy + 1) * HT_W + (ix + 1);
        float win[9];
        #pragma unroll
        for (int dy = 0; dy < 3; ++dy)
          #pragma unroll
          for (int dx = 0; dx < 3; ++dx)
            win[dy*3+dx] = sY1[(n + (dy-1)*HT_W + (dx-1)) * Y1_STRIDE + c];

        float qv[3], kv[3], vv[3];
        #pragma unroll
        for (int kk = 0; kk < 3; ++kk) {
          float aq = 0.f, ak = 0.f, av = 0.f;
          #pragma unroll
          for (int t9 = 0; t9 < 9; ++t9) {
            aq += win[t9] * wq[kk][t9];
            ak += win[t9] * wk[kk][t9];
            av += win[t9] * wv[kk][t9];
          }
          qv[kk] = aq * 0.57735026919f;   // fold 1/sqrt(KS) into q
          kv[kk] = ak; vv[kk] = av;
        }

        // logits[c][d] = sum_k q[c,k] * k[d,k]  (d broadcast from lane d)
        float lg[32];
        #pragma unroll
        for (int d = 0; d < 32; ++d)
          lg[d] = qv[0]*bcastf(kv[0],d) + qv[1]*bcastf(kv[1],d) + qv[2]*bcastf(kv[2],d);

        float m = lg[0];
        #pragma unroll
        for (int d = 1; d < 32; ++d) m = fmaxf(m, lg[d]);
        float s = 0.f;
        #pragma unroll
        for (int d = 0; d < 32; ++d) { float e = __expf(lg[d] - m); lg[d] = e; s += e; }
        const float inv = 1.0f / s;

        float o0 = 0.f, o1 = 0.f, o2 = 0.f;
        #pragma unroll
        for (int d = 0; d < 32; ++d) {
          const float a = lg[d];
          o0 += a * bcastf(vv[0], d);
          o1 += a * bcastf(vv[1], d);
          o2 += a * bcastf(vv[2], d);
        }
        const float yh = (o0*wp[0] + o1*wp[1] + o2*wp[2]) * inv;

        if (h == 0) {
          float pe = 0.f;
          #pragma unroll
          for (int t9 = 0; t9 < 9; ++t9) pe += win[t9] * wpe[t9];
          sY2[n * Y2_STRIDE + c] = pe + yh;
        } else {
          sY2[n * Y2_STRIDE + c] += yh;       // same wave owns pixel n for every head
        }
      }
    }
  }
  __syncthreads();

  // ========== stage 3: outProj GEMM  Out[o][n] = Wout[o][e] Y2[e][n], store ==========
  // M=64 (4 tiles) x N=98 interior (7 tiles) x K=32 (single bf16 wmma step)
  for (int t = wave; t < 28; t += 8) {
    const int Mt = t & 3;
    const int Nt = t >> 2;
    const int nIdx = Nt * 16 + (lane & 15);
    const int iCl  = nIdx < NINT ? nIdx : (NINT - 1);
    const int iy = iCl / TILE_W, ix = iCl % TILE_W;
    const int n  = (iy + 1) * HT_W + (ix + 1);
    const int k0 = 16 * (lane >> 4);

    const float4* bp = (const float4*)(sY2 + n * Y2_STRIDE + k0);
    float4 f0 = bp[0], f1 = bp[1], f2 = bp[2], f3 = bp[3];
    float bf[16] = { f0.x,f0.y,f0.z,f0.w, f1.x,f1.y,f1.z,f1.w,
                     f2.x,f2.y,f2.z,f2.w, f3.x,f3.y,f3.z,f3.w };
    v16bf b;
    #pragma unroll
    for (int q = 0; q < 16; ++q) b[q] = (__bf16)bf[q];

    const int Mrow = Mt * 16 + (lane & 15);
    const int r0 = 16 * (lane >> 4);
    v8bf alo = *(const v8bf*)(sWout + Mrow * E_ + r0);
    v8bf ahi = *(const v8bf*)(sWout + Mrow * E_ + r0 + 8);
    v16bf a = __builtin_shufflevector(alo, ahi, 0,1,2,3,4,5,6,7,8,9,10,11,12,13,14,15);

    v8f acc = {};
    acc = __builtin_amdgcn_wmma_f32_16x16x32_bf16(false, a, false, b, (short)0, acc, false, false);

    if (nIdx < NINT) {
      const int gy = by * TILE_H + iy;
      const int gx = bx * TILE_W + ix;
      #pragma unroll
      for (int r = 0; r < 8; ++r) {
        const int o = Mt * 16 + r + 8 * (lane >> 4);
        out[(((size_t)bz * COUT + o) * IH + gy) * IW + gx] = acc[r];
      }
    }
  }
}

extern "C" void kernel_launch(void* const* d_in, const int* in_sizes, int n_in,
                              void* d_out, int out_size, void* d_ws, size_t ws_size,
                              hipStream_t stream) {
  const float* x     = (const float*)d_in[0];
  const float* w_in  = (const float*)d_in[1];
  const float* w_q   = (const float*)d_in[2];
  const float* w_k   = (const float*)d_in[3];
  const float* w_v   = (const float*)d_in[4];
  const float* w_pe  = (const float*)d_in[5];
  const float* w_p1  = (const float*)d_in[6];
  const float* w_out = (const float*)d_in[7];
  float* o = (float*)d_out;
  (void)in_sizes; (void)n_in; (void)out_size; (void)d_ws; (void)ws_size;

  dim3 grid(IW / TILE_W, IH / TILE_H, 8);   // 4 x 8 x 8 = 256 workgroups
  dim3 block(256);                          // 8 wave32 waves
  MhAConv_fused_kernel<<<grid, block, 0, stream>>>(x, w_in, w_q, w_k, w_v, w_pe, w_p1, w_out, o);
}
